// MultiheadLatentAttention_13615046328899
// MI455X (gfx1250) — compile-verified, matrix operand order checked by reading, not run on hardware
//
#include <hip/hip_runtime.h>
#include <hip/hip_bf16.h>
#include <math.h>

// ---------------------------------------------------------------------------
// MLA forward for gfx1250 (MI455X). All GEMMs + attention go through
// v_wmma_f32_16x16x32_f16 (wave32 WMMA); tile staging of A/K uses the
// Tensor Data Mover (tensor_load_to_lds + s_wait_tensorcnt). f32 inputs are
// converted once to f16; accumulation stays f32. Attention is flash-style.
// ---------------------------------------------------------------------------

typedef __attribute__((ext_vector_type(16))) _Float16 v16h;
typedef __attribute__((ext_vector_type(8)))  _Float16 v8h;
typedef __attribute__((ext_vector_type(8)))  float    v8f;
typedef __attribute__((ext_vector_type(4)))  unsigned int v4u;
typedef __attribute__((ext_vector_type(8)))  int v8i;
typedef __attribute__((ext_vector_type(4)))  int v4i;

#define S_LEN 2048
#define HID   2048
#define NH    32
#define QLR   1536
#define KVLR  512
#define DR    64
#define DN    128
#define DV    128
#define DQK   192                 // DN + DR
#define QDIM  (NH * DQK)          // 6144
#define KVDIM (NH * (DN + DV))    // 8192
#define ODIM  (NH * DV)           // 4096

// Build a 16-half A/B fragment from two aligned 16-byte chunks.
// Per ISA 7.12.2 (16-bit A 16x32): lanes 0-15 hold K {0..7,16..23},
// lanes 16-31 hold K {8..15,24..31}. c0/c1 are hi-half dependent.
__device__ __forceinline__ v16h ld_frag(const _Float16* rowbase, int c0, int c1) {
  v8h lo = *(const v8h*)(rowbase + c0);
  v8h hi = *(const v8h*)(rowbase + c1);
  return __builtin_shufflevector(lo, hi, 0,1,2,3,4,5,6,7,8,9,10,11,12,13,14,15);
}

// ---------------------------------------------------------------------------
// TDM: issue a 2D tensor_load_to_lds of f16 data.
//   lds_off      : byte offset in LDS where the tile starts
//   gaddr        : global address of tile start
//   tile_w       : tile dim0 in elements (row length)
//   tile_h       : tile dim1 (rows)
//   stride       : tensor dim0 stride in elements (row pitch in memory)
//   pad_int/pad_amt : LDS padding codes (ISA 8.4): interval 0=2DW..7=256DW,
//                     amount 0=1DW..127=128DW
// D# packing per CDNA5 ISA §8.3/8.4 (group0 128b, group1 256b; groups 2/3
// zero => 2D tensor). Tensor dims set huge: tiles are always in-bounds.
// Wave-level op (EXEC ignored); call from one wave only per tile.
// ---------------------------------------------------------------------------
__device__ __forceinline__ void tdm_load_2d_f16(
    unsigned lds_off, const void* gaddr, unsigned tile_w, unsigned tile_h,
    unsigned stride, unsigned pad_int, unsigned pad_amt, bool pad_en)
{
  unsigned long long ga = (unsigned long long)(uintptr_t)gaddr;
  v4u g0;
  g0[0] = 1u;                                             // count=1, user mode
  g0[1] = lds_off;                                        // lds_addr
  g0[2] = (unsigned)(ga & 0xffffffffu);                   // global_addr lo
  g0[3] = (unsigned)((ga >> 32) & 0x01ffffffu)            // global_addr[56:32]
        | (2u << 30);                                     // type=2 ("image")
  unsigned w0 = (1u << 16);                               // data_size=1 -> 2B
  if (pad_en) w0 |= (1u << 20) | (pad_int << 22) | (pad_amt << 25);
  const unsigned td0 = 0x40000000u, td1 = 0x40000000u;    // huge tensor dims
  v8i g1;
  g1[0] = (int)w0;                                        // mask=0 | ds | pad
  g1[1] = (int)((td0 & 0xffffu) << 16);                   // abar=0 | td0 lo16
  g1[2] = (int)((td0 >> 16) | ((td1 & 0xffffu) << 16));   // td0 hi | td1 lo
  g1[3] = (int)((td1 >> 16) | (tile_w << 16));            // td1 hi | tile_dim0
  g1[4] = (int)(tile_h & 0xffffu);                        // tile_dim1 | dim2=0
  g1[5] = (int)stride;                                    // dim0_stride lo32
  g1[6] = 0;                                              // stride hi | d1s lo
  g1[7] = 0;
  v4i gz = {0, 0, 0, 0};
#if defined(__clang_major__) && (__clang_major__ >= 23)
  v8i gz8 = {0, 0, 0, 0, 0, 0, 0, 0};
  __builtin_amdgcn_tensor_load_to_lds(g0, g1, gz, gz, gz8, 0);
#else
  __builtin_amdgcn_tensor_load_to_lds(g0, g1, gz, gz, 0);
#endif
}

// ---------------------------------------------------------------------------
// Generic f16 WMMA GEMM: C[M,N] = A[M,K] @ B[K,N], row-major, dims multiples
// of 64/64/32. Block = 64x64 tile, 4 waves; wave w owns rows 16w..16w+15 and
// 4 N-fragments. A tile staged by TDM (wave 0); B tile staged transposed
// ([n][k]) by all threads so B fragment loads are contiguous 16B LDS reads.
// ---------------------------------------------------------------------------
template<bool OUT_F16>
__global__ __launch_bounds__(128) void gemm_f16_wmma(
    const _Float16* __restrict__ A, const _Float16* __restrict__ B,
    void* __restrict__ Cv, int M, int N, int K, int lda, int ldb, int ldc)
{
  __shared__ _Float16 As[64 * 32];
  __shared__ _Float16 Bs[64 * 32];   // [n][k]

  const int tid  = threadIdx.x;
  const int w    = tid >> 5;
  const int lane = tid & 31;
  const int l15  = lane & 15;
  const int hi   = lane >> 4;
  const int c0   = hi * 8;
  const int c1   = hi * 8 + 16;
  const int rowBase = blockIdx.y * 64;
  const int colBase = blockIdx.x * 64;
  const unsigned asOff = (unsigned)(uintptr_t)(&As[0]);

  v8f acc[4] = {};

  for (int k0 = 0; k0 < K; k0 += 32) {
    __syncthreads();   // previous compute done, LDS reusable
    // A tile 64x32 via Tensor Data Mover (async DMA, no VALU/VMEM slots).
    if (w == 0)
      tdm_load_2d_f16(asOff, A + (size_t)rowBase * lda + k0,
                      /*tile_w=*/32, /*tile_h=*/64, /*stride=*/(unsigned)lda,
                      0, 0, false);
    // B tile transposed: Bs[n][k] = B[k0+k][colBase+n] (coalesced reads).
    #pragma unroll
    for (int i = 0; i < 16; ++i) {
      int e = tid + i * 128;
      int k = e >> 6, n = e & 63;
      Bs[n * 32 + k] = B[(size_t)(k0 + k) * ldb + colBase + n];
    }
    if (k0 + 32 < K)
      __builtin_prefetch(B + (size_t)(k0 + 32 + (tid & 31)) * ldb + colBase, 0, 0);
    if (w == 0)
      __builtin_amdgcn_s_wait_tensorcnt(0);
    __syncthreads();

    v16h af = ld_frag(As + (16 * w + l15) * 32, c0, c1);
    #pragma unroll
    for (int f = 0; f < 4; ++f) {
      v16h bf = ld_frag(Bs + (16 * f + l15) * 32, c0, c1);
      acc[f] = __builtin_amdgcn_wmma_f32_16x16x32_f16(
          false, af, false, bf, (short)0, acc[f], false, false);
    }
  }

  // C/D layout: vgpr r, lanes 0-15 -> M=r, lanes 16-31 -> M=r+8, N=l15.
  #pragma unroll
  for (int f = 0; f < 4; ++f)
    #pragma unroll
    for (int r = 0; r < 8; ++r) {
      int row = rowBase + 16 * w + r + hi * 8;
      int col = colBase + 16 * f + l15;
      if (OUT_F16) ((_Float16*)Cv)[(size_t)row * ldc + col] = (_Float16)acc[f][r];
      else         ((float*)Cv)[(size_t)row * ldc + col]    = acc[f][r];
    }
}

// ---------------------------------------------------------------------------
// Elementwise helpers
// ---------------------------------------------------------------------------
__global__ void cvt_f32_f16(const float* __restrict__ s, _Float16* __restrict__ d, int n) {
  int i = blockIdx.x * blockDim.x + threadIdx.x;
  if (i < n) d[i] = (_Float16)s[i];
}

__global__ __launch_bounds__(256) void rmsnorm_f16(
    const float* __restrict__ x, const float* __restrict__ w,
    _Float16* __restrict__ out, int n, int ld_in, int ld_out)
{
  __shared__ float red[256];
  int row = blockIdx.x, tid = threadIdx.x;
  const float* xr = x + (size_t)row * ld_in;
  float s = 0.f;
  for (int i = tid; i < n; i += 256) { float v = xr[i]; s += v * v; }
  red[tid] = s;
  __syncthreads();
  for (int off = 128; off > 0; off >>= 1) {
    if (tid < off) red[tid] += red[tid + off];
    __syncthreads();
  }
  float inv = rsqrtf(red[0] / (float)n + 1e-6f);
  for (int i = tid; i < n; i += 256)
    out[(size_t)row * ld_out + i] = (_Float16)(xr[i] * inv * w[i]);
}

// RoPE on q_rot slices of Q16 [S][6144], in-place (thread owns pair i,i+32).
__global__ void rope_q_inplace(_Float16* __restrict__ q) {
  int idx = blockIdx.x * blockDim.x + threadIdx.x;
  if (idx >= S_LEN * NH * 32) return;
  int i = idx & 31;
  int h = (idx >> 5) & (NH - 1);
  int s = idx >> 10;
  _Float16* base = q + (size_t)s * QDIM + h * DQK + DN;
  float a = (float)base[i], b = (float)base[i + 32];
  float freq = (float)s * __expf(-(float)i * (9.210340371976184f / 32.f)); // 10000^(-i/32)
  float sn, c;
  __sincosf(freq, &sn, &c);
  base[i]      = (_Float16)(a * c - b * sn);
  base[i + 32] = (_Float16)(a * sn + b * c);
}

// RoPE k_rot: reads ckv f32 [S][576] cols 512..575, writes f16 [S][64].
__global__ void rope_k(const float* __restrict__ ckv, _Float16* __restrict__ krot) {
  int idx = blockIdx.x * blockDim.x + threadIdx.x;
  if (idx >= S_LEN * 32) return;
  int i = idx & 31;
  int s = idx >> 5;
  const float* src = ckv + (size_t)s * (KVLR + DR) + KVLR;
  float a = src[i], b = src[i + 32];
  float freq = (float)s * __expf(-(float)i * (9.210340371976184f / 32.f));
  float sn, c;
  __sincosf(freq, &sn, &c);
  _Float16* dst = krot + (size_t)s * DR;
  dst[i]      = (_Float16)(a * c - b * sn);
  dst[i + 32] = (_Float16)(a * sn + b * c);
}

// ---------------------------------------------------------------------------
// Flash attention. Block = (query tile of 64, head); 4 waves, wave w owns
// 16 query rows. K tile Kt[key][192] = [k_pass | k_rot] is assembled by TWO
// TDM descriptors using the LDS-pad feature (k_pass rows: write 64 DW, skip
// 32 DW; k_rot rows: write 32 DW, skip 64 DW). V^T staged manually; P
// round-trips through LDS to re-layout C-frag -> A-frag.
// ---------------------------------------------------------------------------
__global__ __launch_bounds__(128) void mla_attention(
    const _Float16* __restrict__ Q,     // [S][6144] (rot part already roped)
    const _Float16* __restrict__ KV,    // [S][8192] per head: 128 k_pass | 128 v
    const _Float16* __restrict__ KROT,  // [S][64]
    _Float16* __restrict__ AO)          // [S][4096]
{
  __shared__ _Float16 Kt[64 * DQK];   // [key][192]
  __shared__ _Float16 Vt[DV * 64];    // [d][key]  (transposed => B-frag friendly)
  __shared__ _Float16 Pt[64 * 64];    // [qrow][key]

  const int qb   = blockIdx.x;
  const int h    = blockIdx.y;
  const int tid  = threadIdx.x;
  const int w    = tid >> 5;
  const int lane = tid & 31;
  const int l15  = lane & 15;
  const int hi   = lane >> 4;
  const int c0   = hi * 8;
  const int c1   = hi * 8 + 16;
  const unsigned ktOff = (unsigned)(uintptr_t)(&Kt[0]);

  // Preload Q fragments: 16 rows x 192 = 6 A-fragments per wave.
  v16h qf[6];
  {
    int qrow = qb * 64 + 16 * w + l15;
    const _Float16* qbase = Q + (size_t)qrow * QDIM + h * DQK;
    #pragma unroll
    for (int t = 0; t < 6; ++t)
      qf[t] = ld_frag(qbase, 32 * t + c0, 32 * t + c1);
  }

  v8f oacc[8] = {};
  float Mrow[8], Lrow[8];
  #pragma unroll
  for (int r = 0; r < 8; ++r) { Mrow[r] = -1e30f; Lrow[r] = 0.f; }

  const float scale = 0.07216878364870323f;  // 192^-0.5

  for (int kb = 0; kb <= qb; ++kb) {
    __syncthreads();  // previous PV done before restaging
    // Kt via TDM: k_pass rows (128 halves = 64 DW, pad 32 DW) interleaved
    // with k_rot rows (64 halves = 32 DW, pad 64 DW) -> [key][192] layout.
    if (w == 0) {
      tdm_load_2d_f16(ktOff, KV + (size_t)(kb * 64) * KVDIM + h * 256,
                      /*tile_w=*/128, /*tile_h=*/64, /*stride=*/KVDIM,
                      /*pad_int=64DW*/5, /*pad_amt=32DW*/31, true);
      tdm_load_2d_f16(ktOff + DN * 2, KROT + (size_t)(kb * 64) * DR,
                      /*tile_w=*/64, /*tile_h=*/64, /*stride=*/DR,
                      /*pad_int=32DW*/4, /*pad_amt=64DW*/63, true);
    }
    // Stage V transposed: Vt[d][key]. Thread tid owns dim d = tid.
    {
      int d = tid;
      const _Float16* vsrc = KV + (size_t)(kb * 64) * KVDIM + h * 256 + 128 + d;
      #pragma unroll 4
      for (int j = 0; j < 64; ++j)
        Vt[d * 64 + j] = vsrc[(size_t)j * KVDIM];
    }
    if (w == 0)
      __builtin_amdgcn_s_wait_tensorcnt(0);
    __syncthreads();

    // S = Q @ K^T  (reduction over d=192, 6 k-steps, 4 key-fragments)
    v8f sacc[4] = {};
    #pragma unroll
    for (int t = 0; t < 6; ++t) {
      #pragma unroll
      for (int f = 0; f < 4; ++f) {
        v16h bf = ld_frag(Kt + (16 * f + l15) * DQK + 32 * t, c0, c1);
        sacc[f] = __builtin_amdgcn_wmma_f32_16x16x32_f16(
            false, qf[t], false, bf, (short)0, sacc[f], false, false);
      }
    }

    // Scale + causal mask.
    #pragma unroll
    for (int f = 0; f < 4; ++f) {
      int col = kb * 64 + 16 * f + l15;
      #pragma unroll
      for (int r = 0; r < 8; ++r) {
        int row = qb * 64 + 16 * w + r + hi * 8;
        float v = sacc[f][r] * scale;
        sacc[f][r] = (col <= row) ? v : -1e30f;
      }
    }

    // Row max (xor-shuffle masks 1..8 stay inside 16-lane halves: wave32 safe).
    float corr[8];
    #pragma unroll
    for (int r = 0; r < 8; ++r) {
      float m = fmaxf(fmaxf(sacc[0][r], sacc[1][r]), fmaxf(sacc[2][r], sacc[3][r]));
      #pragma unroll
      for (int msk = 1; msk < 16; msk <<= 1)
        m = fmaxf(m, __shfl_xor(m, msk, 32));
      float mnew = fmaxf(Mrow[r], m);
      corr[r] = __expf(Mrow[r] - mnew);
      Mrow[r] = mnew;
    }

    // exp + row sum.
    float rsum[8];
    #pragma unroll
    for (int r = 0; r < 8; ++r) rsum[r] = 0.f;
    #pragma unroll
    for (int f = 0; f < 4; ++f)
      #pragma unroll
      for (int r = 0; r < 8; ++r) {
        float p = __expf(sacc[f][r] - Mrow[r]);
        sacc[f][r] = p;
        rsum[r] += p;
      }
    #pragma unroll
    for (int r = 0; r < 8; ++r) {
      float sg = rsum[r];
      #pragma unroll
      for (int msk = 1; msk < 16; msk <<= 1)
        sg += __shfl_xor(sg, msk, 32);
      Lrow[r] = Lrow[r] * corr[r] + sg;
    }

    // Rescale running O, then spill P to LDS to re-layout into A-fragments.
    #pragma unroll
    for (int f = 0; f < 8; ++f)
      #pragma unroll
      for (int r = 0; r < 8; ++r) oacc[f][r] *= corr[r];
    #pragma unroll
    for (int f = 0; f < 4; ++f)
      #pragma unroll
      for (int r = 0; r < 8; ++r)
        Pt[(16 * w + r + hi * 8) * 64 + 16 * f + l15] = (_Float16)sacc[f][r];
    __syncthreads();

    // O += P @ V  (reduction over 64 keys = 2 k-steps, 8 d-fragments)
    #pragma unroll
    for (int kc = 0; kc < 2; ++kc) {
      v16h pf = ld_frag(Pt + (16 * w + l15) * 64 + kc * 32, c0, c1);
      #pragma unroll
      for (int f = 0; f < 8; ++f) {
        v16h vf = ld_frag(Vt + (16 * f + l15) * 64 + kc * 32, c0, c1);
        oacc[f] = __builtin_amdgcn_wmma_f32_16x16x32_f16(
            false, pf, false, vf, (short)0, oacc[f], false, false);
      }
    }
  }

  // Normalize and write attention output [s][h*128 + d] (f16 for final GEMM).
  #pragma unroll
  for (int f = 0; f < 8; ++f)
    #pragma unroll
    for (int r = 0; r < 8; ++r) {
      int row = qb * 64 + 16 * w + r + hi * 8;
      int col = h * DV + 16 * f + l15;
      AO[(size_t)row * ODIM + col] = (_Float16)(oacc[f][r] / Lrow[r]);
    }
}

// ---------------------------------------------------------------------------
// Host orchestration
// ---------------------------------------------------------------------------
extern "C" void kernel_launch(void* const* d_in, const int* in_sizes, int n_in,
                              void* d_out, int out_size, void* d_ws, size_t ws_size,
                              hipStream_t stream) {
  (void)in_sizes; (void)n_in; (void)out_size; (void)ws_size;
  const float* hidden  = (const float*)d_in[0];
  const float* q_a_w   = (const float*)d_in[1];
  const float* q_a_ln  = (const float*)d_in[2];
  const float* q_b_w   = (const float*)d_in[3];
  const float* kv_a_w  = (const float*)d_in[4];
  const float* kv_a_ln = (const float*)d_in[5];
  const float* kv_b_w  = (const float*)d_in[6];
  const float* o_w     = (const float*)d_in[7];
  float* out = (float*)d_out;

  char* ws = (char*)d_ws;
  size_t off = 0;
  auto alloc = [&](size_t bytes) {
    char* p = ws + off;
    off = (off + bytes + 255) & ~((size_t)255);
    return p;
  };

  _Float16* H16    = (_Float16*)alloc((size_t)S_LEN * HID * 2);
  _Float16* Wqa    = (_Float16*)alloc((size_t)HID * QLR * 2);
  _Float16* Wqb    = (_Float16*)alloc((size_t)QLR * QDIM * 2);
  _Float16* Wkva   = (_Float16*)alloc((size_t)HID * (KVLR + DR) * 2);
  _Float16* Wkvb   = (_Float16*)alloc((size_t)KVLR * KVDIM * 2);
  _Float16* Wo     = (_Float16*)alloc((size_t)ODIM * HID * 2);
  float*    QA32   = (float*)   alloc((size_t)S_LEN * QLR * 4);
  _Float16* QAN16  = (_Float16*)alloc((size_t)S_LEN * QLR * 2);
  _Float16* Q16    = (_Float16*)alloc((size_t)S_LEN * QDIM * 2);
  float*    CKV32  = (float*)   alloc((size_t)S_LEN * (KVLR + DR) * 4);
  _Float16* CKVN16 = (_Float16*)alloc((size_t)S_LEN * KVLR * 2);
  _Float16* KV16   = (_Float16*)alloc((size_t)S_LEN * KVDIM * 2);
  _Float16* KROT16 = (_Float16*)alloc((size_t)S_LEN * DR * 2);
  _Float16* AO16   = (_Float16*)alloc((size_t)S_LEN * ODIM * 2);

  auto cvt = [&](const float* s, _Float16* d, int n) {
    cvt_f32_f16<<<(n + 255) / 256, 256, 0, stream>>>(s, d, n);
  };
  cvt(hidden, H16, S_LEN * HID);
  cvt(q_a_w, Wqa, HID * QLR);
  cvt(q_b_w, Wqb, QLR * QDIM);
  cvt(kv_a_w, Wkva, HID * (KVLR + DR));
  cvt(kv_b_w, Wkvb, KVLR * KVDIM);
  cvt(o_w, Wo, ODIM * HID);

  dim3 blk(128);
  // q_a = hidden @ q_a_w   [2048,1536] f32
  gemm_f16_wmma<false><<<dim3(QLR / 64, S_LEN / 64), blk, 0, stream>>>(
      H16, Wqa, QA32, S_LEN, QLR, HID, HID, QLR, QLR);
  // ckv = hidden @ kv_a_w  [2048,576] f32
  gemm_f16_wmma<false><<<dim3((KVLR + DR) / 64, S_LEN / 64), blk, 0, stream>>>(
      H16, Wkva, CKV32, S_LEN, KVLR + DR, HID, HID, KVLR + DR, KVLR + DR);
  // rms norms
  rmsnorm_f16<<<S_LEN, 256, 0, stream>>>(QA32, q_a_ln, QAN16, QLR, QLR, QLR);
  rmsnorm_f16<<<S_LEN, 256, 0, stream>>>(CKV32, kv_a_ln, CKVN16, KVLR, KVLR + DR, KVLR);
  // q = q_a_norm @ q_b_w   [2048,6144] f16
  gemm_f16_wmma<true><<<dim3(QDIM / 64, S_LEN / 64), blk, 0, stream>>>(
      QAN16, Wqb, Q16, S_LEN, QDIM, QLR, QLR, QDIM, QDIM);
  // kv = ckv_norm @ kv_b_w [2048,8192] f16
  gemm_f16_wmma<true><<<dim3(KVDIM / 64, S_LEN / 64), blk, 0, stream>>>(
      CKVN16, Wkvb, KV16, S_LEN, KVDIM, KVLR, KVLR, KVDIM, KVDIM);
  // rope
  rope_q_inplace<<<(S_LEN * NH * 32 + 255) / 256, 256, 0, stream>>>(Q16);
  rope_k<<<(S_LEN * 32 + 255) / 256, 256, 0, stream>>>(CKV32, KROT16);
  // flash attention
  mla_attention<<<dim3(S_LEN / 64, NH), blk, 0, stream>>>(Q16, KV16, KROT16, AO16);
  // out = attn_out @ o_w   [2048,2048] f32 -> d_out
  gemm_f16_wmma<false><<<dim3(HID / 64, S_LEN / 64), blk, 0, stream>>>(
      AO16, Wo, out, S_LEN, HID, ODIM, ODIM, HID, HID);
}
